// Encoder_29652454211739
// MI455X (gfx1250) — compile-verified, matrix operand order checked by reading, not run on hardware
//
#include <hip/hip_runtime.h>
#include <hip/hip_bf16.h>
#include <math.h>

typedef _Float16 half_t;
typedef __attribute__((ext_vector_type(2)))  _Float16 h2;
typedef __attribute__((ext_vector_type(16))) _Float16 v16h;
typedef __attribute__((ext_vector_type(8)))  float    v8f;

#define B_   2
#define A_   64
#define M_   192
#define P_   30
#define TL_  8
#define T_   264
#define TH_  11
#define H_   8
#define HD_  32
#define LAYERS_ 6
#define NREL_ (B_*T_*T_)   /* 139392 */
#define NTOK_ (B_*T_)      /* 528 */

// ---------------------------------------------------------------------------
// WMMA fragment mapping (wave32, CDNA5 ISA 7.12.2, 16-bit A/B):
//   lane L: m/n = L&15, hf = L>>4
//   VGPR v holds K pair: k = (v&4 ? 16 : 0) + hf*8 + (v&3)*2
//   C/D f32 16x16: VGPR r -> row = r + 8*hf, col = L&15
// B operands are pre-packed so each lane's v16h fragment is 32 contiguous
// bytes: Bp[ ((nt*KT + kt)*32 + lane) * 16 + e ]
// ---------------------------------------------------------------------------
__device__ inline v16h frag_a_lds(const half_t* base, int ldk, int lane) {
  int m = lane & 15, hf = lane >> 4;
  v16h a;
#pragma unroll
  for (int v = 0; v < 8; ++v) {
    int k = ((v & 4) ? 16 : 0) + hf * 8 + (v & 3) * 2;
    h2 p = *(const h2*)(base + m * ldk + k);   // aligned 4B pair
    a[2*v]   = p[0];
    a[2*v+1] = p[1];
  }
  return a;
}

__device__ inline v16h frag_b_packed(const half_t* Bp, int KT, int nt, int kt,
                                     int lane) {
  return *(const v16h*)(Bp + (((size_t)nt * KT + kt) * 32 + lane) * 16);
}

// ---------------------------------------------------------------------------
// Pack one f32 row-major weight W[K,N] into WMMA-fragment-ordered f16 with
// K padded to Kp (multiple of 32).  One thread per (nt,kt,lane).
// ---------------------------------------------------------------------------
__global__ void pack_b_kernel(const float* __restrict__ W,
                              half_t* __restrict__ P, int K, int N, int Kp)
{
  int idx = blockIdx.x * 256 + threadIdx.x;
  int KT = Kp >> 5;
  int total = (N >> 4) * KT * 32;
  if (idx >= total) return;
  int lane = idx & 31;
  int kt = (idx >> 5) % KT;
  int nt = (idx >> 5) / KT;
  int n  = nt * 16 + (lane & 15);
  int hf = lane >> 4;
  v16h out;
#pragma unroll
  for (int v = 0; v < 8; ++v) {
    int k = kt * 32 + ((v & 4) ? 16 : 0) + hf * 8 + (v & 3) * 2;
    out[2*v]   = (k     < K) ? (half_t)W[(size_t)k * N + n]       : (half_t)0.f;
    out[2*v+1] = (k + 1 < K) ? (half_t)W[(size_t)(k + 1) * N + n] : (half_t)0.f;
  }
  *(v16h*)&P[(size_t)idx * 16] = out;
}

// ---------------------------------------------------------------------------
// Generic WMMA GEMM: C[M,N] = act(A[M,K] @ Bp + bias). A f32 row-major,
// Bp fragment-packed f16 (K mult of 32). block = 128 thr (4 waves),
// tile 16 rows x 64 cols.  act: 0 none, 1 relu, 2 exact gelu.
// ---------------------------------------------------------------------------
__global__ __launch_bounds__(128) void wmma_gemm_kernel(
    const float* __restrict__ A, const half_t* __restrict__ Bp,
    const float* __restrict__ bias, float* __restrict__ C,
    int M, int K, int N, int act)
{
  __shared__ alignas(16) half_t As[16 * 1024];
  int tid = threadIdx.x, lane = tid & 31, wave = tid >> 5;
  int rowbase = blockIdx.x * 16;
#pragma unroll 4
  for (int r = 0; r < 16; ++r) {
    int gr = rowbase + r;
    const float* Ar = A + (size_t)gr * K;
    bool ok = gr < M;
    for (int k = tid; k < K; k += 128)
      As[r * K + k] = ok ? (half_t)Ar[k] : (half_t)0.f;
  }
  __syncthreads();
  int nt = blockIdx.y * 4 + wave;
  int KT = K >> 5;
  v8f acc = {};
  for (int kt = 0; kt < KT; ++kt) {
    v16h af = frag_a_lds(&As[kt * 32], K, lane);
    v16h bf = frag_b_packed(Bp, KT, nt, kt, lane);
    acc = __builtin_amdgcn_wmma_f32_16x16x32_f16(false, af, false, bf,
                                                 (short)0, acc, false, false);
  }
  int col = nt * 16 + (lane & 15), hf = lane >> 4;
#pragma unroll
  for (int r = 0; r < 8; ++r) {
    int row = rowbase + r + 8 * hf;
    if (row < M) {
      float v = acc[r] + (bias ? bias[col] : 0.f);
      if (act == 1)      v = fmaxf(v, 0.f);
      else if (act == 2) v = 0.5f * v * (1.f + erff(v * 0.70710678118654752f));
      C[(size_t)row * N + col] = v;
    }
  }
}

// ---------------------------------------------------------------------------
// LN(+ReLU) over 16 rows x 256 cols in LDS. 8 threads/row, butterfly on 8.
// ---------------------------------------------------------------------------
__device__ inline void ln_relu_rows16(const float* __restrict__ ybuf,
                                      half_t* __restrict__ hbuf,
                                      const float* __restrict__ g,
                                      const float* __restrict__ be, int tid)
{
  int row = tid >> 3, sub = tid & 7;
  const float* yr = &ybuf[row * 256];
  float s1 = 0.f, s2 = 0.f;
#pragma unroll 8
  for (int j = 0; j < 32; ++j) {
    float v = yr[sub * 32 + j];
    s1 += v; s2 += v * v;
  }
  for (int o = 4; o >= 1; o >>= 1) {
    s1 += __shfl_xor(s1, o, 8);
    s2 += __shfl_xor(s2, o, 8);
  }
  float mean = s1 * (1.f / 256.f);
  float var  = s2 * (1.f / 256.f) - mean * mean;
  float rstd = rsqrtf(var + 1e-5f);
#pragma unroll 8
  for (int j = 0; j < 32; ++j) {
    int c = sub * 32 + j;
    float v = (yr[c] - mean) * rstd * g[c] + be[c];
    hbuf[row * 256 + c] = (half_t)fmaxf(v, 0.f);
  }
}

// ---------------------------------------------------------------------------
// Fused relation encoder (~100 GFLOP hot spot).  block = 128 thr (4 waves),
// 16 rows x 256 cols per block; all weights fragment-packed f16.
// ---------------------------------------------------------------------------
__global__ __launch_bounds__(128) void rel_encoder_kernel(
    const float* __restrict__ relations, const float* __restrict__ freqs,
    const half_t* __restrict__ w1p,  // 3 x (Kp=160 packed)
    const float* __restrict__ b1,
    const float* __restrict__ g1, const float* __restrict__ be1,
    const half_t* __restrict__ w2p,  // 3 x (256 packed)
    const float* __restrict__ b2,
    const float* __restrict__ go, const float* __restrict__ bo,
    const half_t* __restrict__ wop,  // 256 packed
    const float* __restrict__ bo2,
    half_t* __restrict__ rel_out)
{
  __shared__ alignas(16) half_t feats[16 * 160];  // K=129 padded to 160
  __shared__ alignas(16) half_t hbuf[16 * 256];
  __shared__ float ybuf[16 * 256];
  __shared__ float accbuf[16 * 256];
  int tid = threadIdx.x, lane = tid & 31, wave = tid >> 5;
  int rowbase = blockIdx.x * 16;

  for (int i = tid; i < 16 * 256; i += 128) accbuf[i] = 0.f;

  for (int ch = 0; ch < 3; ++ch) {
    __syncthreads();
    // features: [cos(2*pi*r*f) x64, sin(2*pi*r*f) x64, r, 0-pad]
    for (int i = tid; i < 16 * 160; i += 128) {
      int r = i / 160, k = i - r * 160;
      int g = rowbase + r;
      float rv = (g < NREL_) ? relations[(size_t)g * 3 + ch] : 0.f;
      float val;
      if (k < 64)        val = cosf(6.28318530717958647f * rv * freqs[ch*64 + k]);
      else if (k < 128)  val = sinf(6.28318530717958647f * rv * freqs[ch*64 + k - 64]);
      else if (k == 128) val = rv;
      else               val = 0.f;
      feats[r * 160 + k] = (half_t)val;
    }
    __syncthreads();
    // GEMM1: feats(16x160) @ w1p[ch] + b1 -> ybuf
    const half_t* w1c = w1p + (size_t)ch * 160 * 256;
#pragma unroll
    for (int ct = 0; ct < 4; ++ct) {
      int nt = wave * 4 + ct;
      v8f acc = {};
#pragma unroll
      for (int kt = 0; kt < 5; ++kt) {
        v16h af = frag_a_lds(&feats[kt * 32], 160, lane);
        v16h bf = frag_b_packed(w1c, 5, nt, kt, lane);
        acc = __builtin_amdgcn_wmma_f32_16x16x32_f16(false, af, false, bf,
                                                     (short)0, acc, false, false);
      }
      int col = nt * 16 + (lane & 15), hf = lane >> 4;
#pragma unroll
      for (int r = 0; r < 8; ++r)
        ybuf[(r + 8 * hf) * 256 + col] = acc[r] + b1[ch * 256 + col];
    }
    __syncthreads();
    ln_relu_rows16(ybuf, hbuf, g1 + ch * 256, be1 + ch * 256, tid);
    __syncthreads();
    // GEMM2: h(16x256) @ w2p[ch] + b2 -> accbuf +=
    const half_t* w2c = w2p + (size_t)ch * 256 * 256;
#pragma unroll
    for (int ct = 0; ct < 4; ++ct) {
      int nt = wave * 4 + ct;
      v8f acc = {};
#pragma unroll
      for (int kt = 0; kt < 8; ++kt) {
        v16h af = frag_a_lds(&hbuf[kt * 32], 256, lane);
        v16h bf = frag_b_packed(w2c, 8, nt, kt, lane);
        acc = __builtin_amdgcn_wmma_f32_16x16x32_f16(false, af, false, bf,
                                                     (short)0, acc, false, false);
      }
      int col = nt * 16 + (lane & 15), hf = lane >> 4;
#pragma unroll
      for (int r = 0; r < 8; ++r)
        accbuf[(r + 8 * hf) * 256 + col] += acc[r] + b2[ch * 256 + col];
    }
  }
  __syncthreads();
  ln_relu_rows16(accbuf, hbuf, go, bo, tid);
  __syncthreads();
  // GEMM3: g(16x256) @ wop + bo2 -> rel_out (f16)
#pragma unroll
  for (int ct = 0; ct < 4; ++ct) {
    int nt = wave * 4 + ct;
    v8f acc = {};
#pragma unroll
    for (int kt = 0; kt < 8; ++kt) {
      v16h af = frag_a_lds(&hbuf[kt * 32], 256, lane);
      v16h bf = frag_b_packed(wop, 8, nt, kt, lane);
      acc = __builtin_amdgcn_wmma_f32_16x16x32_f16(false, af, false, bf,
                                                   (short)0, acc, false, false);
    }
    int col = nt * 16 + (lane & 15), hf = lane >> 4;
#pragma unroll
    for (int r = 0; r < 8; ++r) {
      int grow = rowbase + r + 8 * hf;
      if (grow < NREL_)
        rel_out[(size_t)grow * 256 + col] = (half_t)(acc[r] + bo2[col]);
    }
  }
}

// ---------------------------------------------------------------------------
// Attention; one block per (b,t); wave per head (wave32).
// ---------------------------------------------------------------------------
__global__ __launch_bounds__(256) void attn_kernel(
    const float* __restrict__ qkv, const half_t* __restrict__ rel,
    const float* __restrict__ maskb, float* __restrict__ attnv)
{
  int row = blockIdx.x;            // b*T + t
  int b = row / T_;
  int tid = threadIdx.x;
  __shared__ float qs[H_ * HD_];
  __shared__ float sc[H_][T_];
  {
    int h = tid >> 5, d = tid & 31;
    qs[tid] = qkv[(size_t)row * 768 + h * 96 + d];
  }
  __syncthreads();
  const float scale = 0.17677669529663687f;  // 1/sqrt(32)
  // reference bias broadcasts over keys (query-indexed) -> row constant
  float qbias = maskb[row];
  for (int idx = tid; idx < H_ * T_; idx += 256) {
    int h = idx / T_, s = idx - h * T_;
    const float*  kr = &qkv[(size_t)(b * T_ + s) * 768 + h * 96 + 32];
    const half_t* rr = &rel[((size_t)row * T_ + s) * 256 + h * 32];
    float acc = 0.f;
#pragma unroll 8
    for (int d = 0; d < 32; ++d) acc += qs[h * 32 + d] * (kr[d] + (float)rr[d]);
    sc[h][s] = acc * scale - qbias;
  }
  __syncthreads();
  int h = tid >> 5, ln = tid & 31;
  float mx = -1e30f;
  for (int s = ln; s < T_; s += 32) mx = fmaxf(mx, sc[h][s]);
  for (int o = 16; o >= 1; o >>= 1) mx = fmaxf(mx, __shfl_xor(mx, o, 32));
  float sm = 0.f;
  for (int s = ln; s < T_; s += 32) {
    float e = __expf(sc[h][s] - mx); sc[h][s] = e; sm += e;
  }
  for (int o = 16; o >= 1; o >>= 1) sm += __shfl_xor(sm, o, 32);
  float inv = 1.f / sm;
  __syncthreads();
  {
    int d = ln;
    float acc = 0.f;
    for (int s = 0; s < T_; ++s) {
      float p = sc[h][s] * inv;
      acc += p * (qkv[(size_t)(b * T_ + s) * 768 + h * 96 + 64 + d] +
                  (float)rel[((size_t)row * T_ + s) * 256 + h * 32 + d]);
    }
    attnv[(size_t)row * 256 + h * 32 + d] = acc;
  }
}

// ---------------------------------------------------------------------------
// out = LN(a + r) * g + be   (row-wise over 256)
// ---------------------------------------------------------------------------
__global__ __launch_bounds__(256) void add_ln_kernel(
    const float* __restrict__ a, const float* __restrict__ r,
    const float* __restrict__ g, const float* __restrict__ be,
    float* __restrict__ out)
{
  int row = blockIdx.x, tid = threadIdx.x;
  float v = a[(size_t)row * 256 + tid] + r[(size_t)row * 256 + tid];
  __shared__ float r1[8], r2[8];
  float s1 = v, s2 = v * v;
  for (int o = 16; o >= 1; o >>= 1) {
    s1 += __shfl_xor(s1, o, 32);
    s2 += __shfl_xor(s2, o, 32);
  }
  if ((tid & 31) == 0) { r1[tid >> 5] = s1; r2[tid >> 5] = s2; }
  __syncthreads();
  float t1 = 0.f, t2 = 0.f;
#pragma unroll
  for (int i = 0; i < 8; ++i) { t1 += r1[i]; t2 += r2[i]; }
  float mean = t1 * (1.f / 256.f);
  float var  = t2 * (1.f / 256.f) - mean * mean;
  float rstd = rsqrtf(var + 1e-5f);
  out[(size_t)row * 256 + tid] = (v - mean) * rstd * g[tid] + be[tid];
}

// ---------------------------------------------------------------------------
// Input encoders (tiny)
// ---------------------------------------------------------------------------
__global__ __launch_bounds__(256) void encode_agents_kernel(
    const float* __restrict__ hist, const int* __restrict__ atype,
    const float* __restrict__ b1, const float* __restrict__ b2,
    const float* __restrict__ b3, const float* __restrict__ temb,
    const float* __restrict__ w1, const float* __restrict__ w2,
    const float* __restrict__ w3, float* __restrict__ x)
{
  int blk = blockIdx.x, b = blk / A_, a = blk - b * A_;
  int tid = threadIdx.x;
  __shared__ float cur[5];
  __shared__ float h1[128];
  __shared__ float h2s[256];
  if (tid < 5) cur[tid] = hist[((size_t)(b * A_ + a) * TH_ + (TH_ - 1)) * 8 + 3 + tid];
  __syncthreads();
  if (tid < 128) {
    float s = b1[tid];
#pragma unroll
    for (int i = 0; i < 5; ++i) s += cur[i] * w1[i * 128 + tid];
    h1[tid] = fmaxf(s, 0.f);
  }
  __syncthreads();
  {
    float s = b2[tid];
    for (int i = 0; i < 128; ++i) s += h1[i] * w2[i * 256 + tid];
    h2s[tid] = fmaxf(s, 0.f);
  }
  __syncthreads();
  float s = b3[tid];
  for (int i = 0; i < 256; ++i) s += h2s[i] * w3[i * 256 + tid];
  int ty = atype[b * A_ + a];
  x[((size_t)b * T_ + a) * 256 + tid] = s + temb[ty * 256 + tid];
}

__global__ __launch_bounds__(256) void encode_map_kernel(
    const float* __restrict__ poly,
    const float* __restrict__ b1, const float* __restrict__ b2,
    const float* __restrict__ tlemb, const float* __restrict__ temb,
    const float* __restrict__ w1, const float* __restrict__ w2,
    float* __restrict__ x)
{
  int blk = blockIdx.x, b = blk / M_, m = blk - b * M_;
  int tid = threadIdx.x;
  __shared__ float h1[128];
  const float* pb = &poly[(size_t)(b * M_ + m) * P_ * 5];
  float pooled = -1e30f;
  for (int p = 0; p < P_; ++p) {
    __syncthreads();
    if (tid < 128) {
      float s = b1[tid];
      s += pb[p * 5 + 0] * w1[0 * 128 + tid];
      s += pb[p * 5 + 1] * w1[1 * 128 + tid];
      s += pb[p * 5 + 2] * w1[2 * 128 + tid];
      h1[tid] = fmaxf(s, 0.f);
    }
    __syncthreads();
    float s = b2[tid];
    for (int i = 0; i < 128; ++i) s += h1[i] * w2[i * 256 + tid];
    pooled = fmaxf(pooled, s);
  }
  int tlt = (int)pb[3]; tlt = tlt < 0 ? 0 : (tlt > 7 ? 7 : tlt);
  int pt  = (int)pb[4]; pt  = pt  < 0 ? 0 : (pt  > 20 ? 20 : pt);
  x[((size_t)b * T_ + A_ + m) * 256 + tid] =
      pooled + tlemb[tlt * 256 + tid] + temb[pt * 256 + tid];
}

__global__ __launch_bounds__(256) void encode_tl_kernel(
    const float* __restrict__ tlp, const float* __restrict__ temb,
    float* __restrict__ x)
{
  int blk = blockIdx.x, b = blk / TL_, t = blk - b * TL_;
  int tid = threadIdx.x;
  int lt = (int)tlp[(b * TL_ + t) * 3 + 2];
  lt = lt < 0 ? 0 : (lt > 7 ? 7 : lt);
  x[((size_t)b * T_ + A_ + M_ + t) * 256 + tid] = temb[lt * 256 + tid];
}

__global__ void mask_kernel(const int* __restrict__ interested,
                            const int* __restrict__ pvalid,
                            const float* __restrict__ tlp,
                            float* __restrict__ maskb)
{
  int idx = blockIdx.x * blockDim.x + threadIdx.x;
  if (idx >= NTOK_) return;
  int b = idx / T_, j = idx - b * T_;
  float m;
  if (j < A_)            m = (interested[b * A_ + j] == 0) ? 1.f : 0.f;
  else if (j < A_ + M_)  m = (pvalid[b * M_ + (j - A_)] == 0) ? 1.f : 0.f;
  else {
    int t = j - A_ - M_;
    float s = tlp[(b * TL_ + t) * 3 + 0] + tlp[(b * TL_ + t) * 3 + 1] +
              tlp[(b * TL_ + t) * 3 + 2];
    m = (s == 0.f) ? 1.f : 0.f;
  }
  maskb[idx] = m * 1e9f;
}

// ---------------------------------------------------------------------------
// Host launcher
// ---------------------------------------------------------------------------
static inline int pack_blocks(int K_pad, int N) {
  int total = (N >> 4) * (K_pad >> 5) * 32;
  return (total + 255) / 256;
}

extern "C" void kernel_launch(void* const* d_in, const int* in_sizes, int n_in,
                              void* d_out, int out_size, void* d_ws,
                              size_t ws_size, hipStream_t stream)
{
  (void)in_sizes; (void)n_in; (void)out_size; (void)ws_size;
  const float* agents_history = (const float*)d_in[0];
  const int*   agents_type    = (const int*)  d_in[1];
  const int*   agents_inter   = (const int*)  d_in[2];
  const float* polylines      = (const float*)d_in[3];
  const int*   poly_valid     = (const int*)  d_in[4];
  const float* tl_points      = (const float*)d_in[5];
  const float* relations      = (const float*)d_in[6];
  // params (jax pytree order: dict keys sorted alphabetically)
  const float* ag_b1   = (const float*)d_in[7];
  const float* ag_b2   = (const float*)d_in[8];
  const float* ag_b3   = (const float*)d_in[9];
  const float* ag_temb = (const float*)d_in[10];
  const float* ag_w1   = (const float*)d_in[11];
  const float* ag_w2   = (const float*)d_in[12];
  const float* ag_w3   = (const float*)d_in[13];
  const float* mp_b1    = (const float*)d_in[14];
  const float* mp_b2    = (const float*)d_in[15];
  const float* mp_tlemb = (const float*)d_in[16];
  const float* mp_temb  = (const float*)d_in[17];
  const float* mp_w1    = (const float*)d_in[18];
  const float* mp_w2    = (const float*)d_in[19];
  const float* rl_b1    = (const float*)d_in[20];
  const float* rl_b2    = (const float*)d_in[21];
  const float* rl_be1   = (const float*)d_in[22];
  const float* rl_bo    = (const float*)d_in[23];
  const float* rl_bo2   = (const float*)d_in[24];
  const float* rl_freqs = (const float*)d_in[25];
  const float* rl_g1    = (const float*)d_in[26];
  const float* rl_go    = (const float*)d_in[27];
  const float* rl_w1    = (const float*)d_in[28];
  const float* rl_w2    = (const float*)d_in[29];
  const float* rl_wo    = (const float*)d_in[30];
  const float* tf_be1  = (const float*)d_in[31];
  const float* tf_be2  = (const float*)d_in[32];
  const float* tf_bf1  = (const float*)d_in[33];
  const float* tf_bf2  = (const float*)d_in[34];
  const float* tf_bin  = (const float*)d_in[35];
  const float* tf_bout = (const float*)d_in[36];
  const float* tf_g1   = (const float*)d_in[37];
  const float* tf_g2   = (const float*)d_in[38];
  const float* tf_wf1  = (const float*)d_in[39];
  const float* tf_wf2  = (const float*)d_in[40];
  const float* tf_win  = (const float*)d_in[41];
  const float* tf_wout = (const float*)d_in[42];
  const float* tl_temb = (const float*)d_in[43];

  // ---- workspace layout ----
  char* ws = (char*)d_ws;
  size_t off = 0;
  half_t* relbuf = (half_t*)(ws + off); off += (size_t)NREL_ * 256 * sizeof(half_t);
  float*  x      = (float*)(ws + off);  off += (size_t)NTOK_ * 256 * sizeof(float);
  float*  x1     = (float*)(ws + off);  off += (size_t)NTOK_ * 256 * sizeof(float);
  float*  tmp    = (float*)(ws + off);  off += (size_t)NTOK_ * 256 * sizeof(float);
  float*  attnv  = (float*)(ws + off);  off += (size_t)NTOK_ * 256 * sizeof(float);
  float*  qkvb   = (float*)(ws + off);  off += (size_t)NTOK_ * 768 * sizeof(float);
  float*  ffb    = (float*)(ws + off);  off += (size_t)NTOK_ * 1024 * sizeof(float);
  float*  maskb  = (float*)(ws + off);  off += (size_t)NTOK_ * sizeof(float);
  off = (off + 63) & ~(size_t)63;
  half_t* w1p  = (half_t*)(ws + off); off += (size_t)3 * 160 * 256  * sizeof(half_t);
  half_t* w2p  = (half_t*)(ws + off); off += (size_t)3 * 256 * 256  * sizeof(half_t);
  half_t* wop  = (half_t*)(ws + off); off += (size_t)256 * 256      * sizeof(half_t);
  half_t* winp = (half_t*)(ws + off); off += (size_t)LAYERS_ * 256 * 768  * sizeof(half_t);
  half_t* woutp= (half_t*)(ws + off); off += (size_t)LAYERS_ * 256 * 256  * sizeof(half_t);
  half_t* wf1p = (half_t*)(ws + off); off += (size_t)LAYERS_ * 256 * 1024 * sizeof(half_t);
  half_t* wf2p = (half_t*)(ws + off); off += (size_t)LAYERS_ * 1024 * 256 * sizeof(half_t);

  // ---- pack all GEMM B operands into WMMA fragment order (f16) ----
  for (int ch = 0; ch < 3; ++ch) {
    pack_b_kernel<<<pack_blocks(160, 256), 256, 0, stream>>>(
        rl_w1 + (size_t)ch * 129 * 256, w1p + (size_t)ch * 160 * 256, 129, 256, 160);
    pack_b_kernel<<<pack_blocks(256, 256), 256, 0, stream>>>(
        rl_w2 + (size_t)ch * 256 * 256, w2p + (size_t)ch * 256 * 256, 256, 256, 256);
  }
  pack_b_kernel<<<pack_blocks(256, 256), 256, 0, stream>>>(rl_wo, wop, 256, 256, 256);
  for (int l = 0; l < LAYERS_; ++l) {
    pack_b_kernel<<<pack_blocks(256, 768), 256, 0, stream>>>(
        tf_win + (size_t)l * 256 * 768, winp + (size_t)l * 256 * 768, 256, 768, 256);
    pack_b_kernel<<<pack_blocks(256, 256), 256, 0, stream>>>(
        tf_wout + (size_t)l * 256 * 256, woutp + (size_t)l * 256 * 256, 256, 256, 256);
    pack_b_kernel<<<pack_blocks(256, 1024), 256, 0, stream>>>(
        tf_wf1 + (size_t)l * 256 * 1024, wf1p + (size_t)l * 256 * 1024, 256, 1024, 256);
    pack_b_kernel<<<pack_blocks(1024, 256), 256, 0, stream>>>(
        tf_wf2 + (size_t)l * 1024 * 256, wf2p + (size_t)l * 1024 * 256, 1024, 256, 1024);
  }

  // ---- encoders ----
  encode_agents_kernel<<<B_ * A_, 256, 0, stream>>>(
      agents_history, agents_type, ag_b1, ag_b2, ag_b3, ag_temb,
      ag_w1, ag_w2, ag_w3, x);
  encode_map_kernel<<<B_ * M_, 256, 0, stream>>>(
      polylines, mp_b1, mp_b2, mp_tlemb, mp_temb, mp_w1, mp_w2, x);
  encode_tl_kernel<<<B_ * TL_, 256, 0, stream>>>(tl_points, tl_temb, x);
  mask_kernel<<<(NTOK_ + 255) / 256, 256, 0, stream>>>(
      agents_inter, poly_valid, tl_points, maskb);

  // ---- relation encoder (dominant WMMA workload) ----
  rel_encoder_kernel<<<NREL_ / 16, 128, 0, stream>>>(
      relations, rl_freqs, w1p, rl_b1, rl_g1, rl_be1,
      w2p, rl_b2, rl_go, rl_bo, wop, rl_bo2, relbuf);

  // ---- transformer layers ----
  for (int l = 0; l < LAYERS_; ++l) {
    const half_t* win  = winp  + (size_t)l * 256 * 768;
    const float*  bin  = tf_bin  + (size_t)l * 768;
    const half_t* wout = woutp + (size_t)l * 256 * 256;
    const float*  bout = tf_bout + (size_t)l * 256;
    const float*  g1   = tf_g1   + (size_t)l * 256;
    const float*  be1  = tf_be1  + (size_t)l * 256;
    const float*  g2   = tf_g2   + (size_t)l * 256;
    const float*  be2  = tf_be2  + (size_t)l * 256;
    const half_t* wf1  = wf1p  + (size_t)l * 256 * 1024;
    const float*  bf1  = tf_bf1  + (size_t)l * 1024;
    const half_t* wf2  = wf2p  + (size_t)l * 1024 * 256;
    const float*  bf2  = tf_bf2  + (size_t)l * 256;

    wmma_gemm_kernel<<<dim3(NTOK_ / 16, 768 / 64), 128, 0, stream>>>(
        x, win, bin, qkvb, NTOK_, 256, 768, 0);
    attn_kernel<<<NTOK_, 256, 0, stream>>>(qkvb, relbuf, maskb, attnv);
    wmma_gemm_kernel<<<dim3(NTOK_ / 16, 256 / 64), 128, 0, stream>>>(
        attnv, wout, bout, tmp, NTOK_, 256, 256, 0);
    add_ln_kernel<<<NTOK_, 256, 0, stream>>>(tmp, x, g1, be1, x1);
    wmma_gemm_kernel<<<dim3(NTOK_ / 16, 1024 / 64), 128, 0, stream>>>(
        x1, wf1, bf1, ffb, NTOK_, 256, 1024, 2);
    wmma_gemm_kernel<<<dim3(NTOK_ / 16, 256 / 64), 128, 0, stream>>>(
        ffb, wf2, bf2, tmp, NTOK_, 1024, 256, 0);
    float* outp = (l == LAYERS_ - 1) ? (float*)d_out : x;
    add_ln_kernel<<<NTOK_, 256, 0, stream>>>(tmp, x1, g2, be2, outp);
  }
}
